// CausalSelfAttention_8615704396110
// MI455X (gfx1250) — matrix operand order repaired and checked by reading, not compiled, not run on hardware
//
#include <hip/hip_runtime.h>
#include <hip/hip_bf16.h>

typedef __bf16 bf16;
typedef __attribute__((ext_vector_type(16))) bf16  v16bf;
typedef __attribute__((ext_vector_type(8)))  bf16  v8bf;
typedef __attribute__((ext_vector_type(8)))  float v8f;

#define B_    2
#define T_    2048
#define C_    1024
#define NH_   16
#define NKV_  4
#define HD_   64
#define NQKV_ 1536   // (NH + 2*NKV) * HD

// ---------- helpers ----------

static __device__ __forceinline__ bf16 f2bf(float f) {
  unsigned u = __builtin_bit_cast(unsigned, f);
  unsigned r = u + 0x7FFFu + ((u >> 16) & 1u);   // round-to-nearest-even
  unsigned short h = (unsigned short)(r >> 16);
  return __builtin_bit_cast(bf16, h);
}

// CDNA5 async global->LDS copy of 16 bytes (one instruction per wave,
// each lane copies 16B). Tracked by ASYNCcnt.
static __device__ __forceinline__ void async_copy16(void* lds_dst, const void* gsrc) {
  unsigned lds = (unsigned)(size_t)lds_dst;           // low 32 bits = LDS offset
  unsigned long long ga = (unsigned long long)(size_t)gsrc;
  asm volatile("global_load_async_to_lds_b128 %0, %1, off"
               :: "v"(lds), "v"(ga) : "memory");
}
static __device__ __forceinline__ void async_wait0() {
  asm volatile("s_wait_asynccnt 0" ::: "memory");
}

// A-fragment (16x32 bf16, M x K): lane holds row m=lane%16, half=lane/16.
// elems 0..7  -> k = half*8 + e ; elems 8..15 -> k = 16 + half*8 + e
static __device__ __forceinline__ v16bf load_frag_a(const bf16* base, int ld) {
  int lane = threadIdx.x & 31;
  int m = lane & 15, half = lane >> 4;
  v8bf lo = *(const v8bf*)(base + m * ld + half * 8);
  v8bf hi = *(const v8bf*)(base + m * ld + 16 + half * 8);
  v16bf r;
#pragma unroll
  for (int i = 0; i < 8; ++i) { r[i] = lo[i]; r[i + 8] = hi[i]; }
  return r;
}

// B-fragment (32x16 bf16, K x N) from LDS stored transposed: base[n*ld+k]=B[k][n].
// Lane holds col n=lane%16, kh=lane/16, elems 0..7 -> k = kh*8 + e; elems 8..15 -> k = 16 + kh*8 + e.
static __device__ __forceinline__ v16bf load_frag_b(const bf16* base, int ld) {
  int lane = threadIdx.x & 31;
  int n = lane & 15, kh = lane >> 4;
  v8bf lo = *(const v8bf*)(base + n * ld + kh * 8);
  v8bf hi = *(const v8bf*)(base + n * ld + 16 + kh * 8);
  v16bf r;
#pragma unroll
  for (int i = 0; i < 8; ++i) { r[i] = lo[i]; r[i + 8] = hi[i]; }
  return r;
}

// ---------- kernel 1a: f32 -> bf16 convert ----------

__global__ __launch_bounds__(256) void cvt_f32_bf16(const float* __restrict__ in,
                                                    bf16* __restrict__ out, int n) {
  int i = blockIdx.x * 256 + threadIdx.x;
  if (i < n) out[i] = f2bf(in[i]);
}

// ---------- kernel 1b: f32 -> bf16 convert + transpose (weights) ----------
// in: [K][N] row-major f32; out: [N][K] row-major bf16.

__global__ __launch_bounds__(256) void cvt_transpose_bf16(const float* __restrict__ in,
                                                          bf16* __restrict__ out,
                                                          int K, int N) {
  int idx = blockIdx.x * 256 + threadIdx.x;
  if (idx >= K * N) return;
  int k = idx / N, n = idx % N;      // coalesced read
  out[(size_t)n * K + k] = f2bf(in[idx]);
}

// ---------- kernel 2: bf16 GEMM, fp32 out: C[MxN] = A[MxK] * Bt[NxK]^T ----------
// Both operands K-major; all LDS staging via async b128 copies.

__global__ __launch_bounds__(256) void gemm_bf16(const bf16* __restrict__ A,
                                                 const bf16* __restrict__ Bt,
                                                 float* __restrict__ C,
                                                 int M, int N, int K) {
  __shared__ __attribute__((aligned(16))) bf16 sA[128 * 40];  // [m][k], ld=40
  __shared__ __attribute__((aligned(16))) bf16 sB[128 * 40];  // [n][k], ld=40

  const int tid = threadIdx.x;
  const int m0 = blockIdx.y * 128, n0 = blockIdx.x * 128;
  const int w = tid >> 5, wr = w >> 2, wc = w & 3;  // 2 x 4 wave grid
  const int lane = tid & 31, n = lane & 15, half = lane >> 4;

  v8f acc[4][2];
#pragma unroll
  for (int mi = 0; mi < 4; ++mi)
#pragma unroll
    for (int ni = 0; ni < 2; ++ni)
#pragma unroll
      for (int e = 0; e < 8; ++e) acc[mi][ni][e] = 0.0f;

  for (int kk = 0; kk < K; kk += 32) {
    // stage A and B tiles: 128 rows x 32 k each, async 16B copies
#pragma unroll
    for (int i = 0; i < 2; ++i) {
      int lin = tid + i * 256;
      int r = lin >> 2, ks = (lin & 3) * 8;
      async_copy16(sA + r * 40 + ks, A + (size_t)(m0 + r) * K + kk + ks);
      async_copy16(sB + r * 40 + ks, Bt + (size_t)(n0 + r) * K + kk + ks);
    }
    async_wait0();
    __syncthreads();

    v16bf af[4], bfm[2];
#pragma unroll
    for (int mi = 0; mi < 4; ++mi)
      af[mi] = load_frag_a(sA + (wr * 64 + mi * 16) * 40, 40);
#pragma unroll
    for (int ni = 0; ni < 2; ++ni)
      bfm[ni] = load_frag_b(sB + (wc * 32 + ni * 16) * 40, 40);

#pragma unroll
    for (int mi = 0; mi < 4; ++mi)
#pragma unroll
      for (int ni = 0; ni < 2; ++ni)
        acc[mi][ni] = __builtin_amdgcn_wmma_f32_16x16x32_bf16(
            false, af[mi], false, bfm[ni], (short)0, acc[mi][ni], false, false);
    __syncthreads();
  }

#pragma unroll
  for (int mi = 0; mi < 4; ++mi)
#pragma unroll
    for (int ni = 0; ni < 2; ++ni) {
      float* pc = C + (size_t)(m0 + wr * 64 + mi * 16 + 8 * half) * N
                    + (n0 + wc * 32 + ni * 16 + n);
#pragma unroll
      for (int e = 0; e < 8; ++e) pc[(size_t)e * N] = acc[mi][ni][e];
    }
}

// ---------- kernel 3: RoPE + scatter qkv(f32) -> Q/K head-major, V d-major ----------

__global__ __launch_bounds__(256) void rope_scatter(const float* __restrict__ qkv,
                                                    bf16* __restrict__ Q,
                                                    bf16* __restrict__ Kd,
                                                    bf16* __restrict__ Vt) {
  const int NPAIR = NQKV_ / 2;
  int idx = blockIdx.x * 256 + threadIdx.x;
  if (idx >= B_ * T_ * NPAIR) return;
  int p = idx % NPAIR;
  int row = idx / NPAIR;
  int t = row % T_, b = row / T_;
  int col = p * 2;
  float v0 = qkv[(size_t)row * NQKV_ + col];
  float v1 = qkv[(size_t)row * NQKV_ + col + 1];

  if (col < NH_ * HD_) {                       // Q with RoPE, [b][h][t][d]
    int h = col >> 6, d = col & 63;
    float ang = (float)t * __powf(10000.0f, -(float)(d >> 1) / 32.0f);
    float s, c;
    __sincosf(ang, &s, &c);
    size_t o = (((size_t)b * NH_ + h) * T_ + t) * HD_ + d;
    Q[o]     = f2bf(v0 * c - v1 * s);
    Q[o + 1] = f2bf(v0 * s + v1 * c);
  } else if (col < (NH_ + NKV_) * HD_) {       // K with RoPE, [b][h][t][d]
    int cc = col - NH_ * HD_;
    int h = cc >> 6, d = cc & 63;
    float ang = (float)t * __powf(10000.0f, -(float)(d >> 1) / 32.0f);
    float s, c;
    __sincosf(ang, &s, &c);
    size_t o = (((size_t)b * NKV_ + h) * T_ + t) * HD_ + d;
    Kd[o]     = f2bf(v0 * c - v1 * s);
    Kd[o + 1] = f2bf(v0 * s + v1 * c);
  } else {                                     // V copy, transposed: [b][h][d][t]
    int cc = col - (NH_ + NKV_) * HD_;
    int h = cc >> 6, d = cc & 63;
    size_t o = (((size_t)b * NKV_ + h) * HD_ + d) * T_ + t;
    Vt[o]      = f2bf(v0);     // dim d
    Vt[o + T_] = f2bf(v1);     // dim d+1
  }
}

// ---------- kernel 4: flash attention (64-query tile per block, 4 waves) ----------

__global__ __launch_bounds__(128) void attn_flash(const bf16* __restrict__ Qg,
                                                  const bf16* __restrict__ Kg,
                                                  const bf16* __restrict__ Vtg,
                                                  bf16* __restrict__ Yg) {
  __shared__ __attribute__((aligned(16))) bf16 sQ[64 * 72];   // [q][d]
  __shared__ __attribute__((aligned(16))) bf16 sK[64 * 72];   // [j][d] (B-layout for S)
  __shared__ __attribute__((aligned(16))) bf16 sVT[64 * 72];  // [d][j] (B-layout for O)
  __shared__ __attribute__((aligned(16))) bf16 sP[64 * 72];   // [q][j] probs (A-layout)
  __shared__ __attribute__((aligned(16))) float sS[64 * 68];  // [q][j] scores
  __shared__ float row_m[64], row_l[64], row_al[64];

  const int tid = threadIdx.x;
  const int w = tid >> 5;
  const int lane = tid & 31, n = lane & 15, half = lane >> 4;
  const int qt = blockIdx.x, h = blockIdx.y, b = blockIdx.z;
  const int hkv = h / (NH_ / NKV_);

  const bf16* Qh  = Qg  + ((size_t)b * NH_ + h) * T_ * HD_;
  const bf16* Kh  = Kg  + ((size_t)b * NKV_ + hkv) * T_ * HD_;
  const bf16* Vth = Vtg + ((size_t)b * NKV_ + hkv) * HD_ * T_;  // [d][t]

  // stage Q tile (64 x 64) async
#pragma unroll
  for (int i = 0; i < 4; ++i) {
    int lin = tid + i * 128;
    int m = lin >> 3, ds = (lin & 7) * 8;
    async_copy16(sQ + m * 72 + ds, Qh + (size_t)(qt * 64 + m) * HD_ + ds);
  }
  if (tid < 64) { row_m[tid] = -1e30f; row_l[tid] = 0.0f; }
  async_wait0();
  __syncthreads();

  v16bf qa[2];
  qa[0] = load_frag_a(sQ + (w * 16) * 72, 72);       // k = 0..31
  qa[1] = load_frag_a(sQ + (w * 16) * 72 + 32, 72);  // k = 32..63

  v8f o[4];
#pragma unroll
  for (int ni = 0; ni < 4; ++ni)
#pragma unroll
    for (int e = 0; e < 8; ++e) o[ni][e] = 0.0f;

  for (int jt = 0; jt <= qt; ++jt) {
    __syncthreads();  // previous iteration's reads of sK/sVT/sP complete
    // stage K rows and V^T rows, all contiguous 16B async copies
#pragma unroll
    for (int i = 0; i < 4; ++i) {
      int lin = tid + i * 128;
      int r = lin >> 3, ds = (lin & 7) * 8;
      async_copy16(sK + r * 72 + ds, Kh + (size_t)(jt * 64 + r) * HD_ + ds);
      // r = d-dim row of V^T, ds = key offset within tile
      async_copy16(sVT + r * 72 + ds, Vth + (size_t)r * T_ + jt * 64 + ds);
    }
    async_wait0();
    __syncthreads();

    // S = Q * K^T  (wave w: rows w*16..+15, all 64 key cols)
#pragma unroll
    for (int ni = 0; ni < 4; ++ni) {
      v8f s;
#pragma unroll
      for (int e = 0; e < 8; ++e) s[e] = 0.0f;
      v16bf kb0 = load_frag_b(sK + (ni * 16) * 72, 72);
      v16bf kb1 = load_frag_b(sK + (ni * 16) * 72 + 32, 72);
      s = __builtin_amdgcn_wmma_f32_16x16x32_bf16(false, qa[0], false, kb0,
                                                  (short)0, s, false, false);
      s = __builtin_amdgcn_wmma_f32_16x16x32_bf16(false, qa[1], false, kb1,
                                                  (short)0, s, false, false);
#pragma unroll
      for (int e = 0; e < 8; ++e) {
        int m = e + 8 * half;
        int qrow = qt * 64 + w * 16 + m;
        int jcol = jt * 64 + ni * 16 + n;
        float v = s[e] * 0.125f;       // 1/sqrt(64)
        if (jcol > qrow) v = -1e30f;   // causal mask
        sS[(w * 16 + m) * 68 + ni * 16 + n] = v;
      }
    }
    __syncthreads();

    // online softmax: one thread per query row
    if (tid < 64) {
      float mx = row_m[tid];
      float rmax = -1e30f;
#pragma unroll 8
      for (int j = 0; j < 64; ++j) rmax = fmaxf(rmax, sS[tid * 68 + j]);
      float mnew = fmaxf(mx, rmax);
      float alpha = __expf(mx - mnew);
      float l = row_l[tid] * alpha;
#pragma unroll 8
      for (int j = 0; j < 64; ++j) {
        float pv = __expf(sS[tid * 68 + j] - mnew);
        l += pv;
        sP[tid * 72 + j] = f2bf(pv);
      }
      row_m[tid] = mnew;
      row_l[tid] = l;
      row_al[tid] = alpha;
    }
    __syncthreads();

    // rescale O and accumulate P * V
    float al[8];
#pragma unroll
    for (int e = 0; e < 8; ++e) al[e] = row_al[w * 16 + e + 8 * half];
#pragma unroll
    for (int ni = 0; ni < 4; ++ni)
#pragma unroll
      for (int e = 0; e < 8; ++e) o[ni][e] *= al[e];

    v16bf pa0 = load_frag_a(sP + (w * 16) * 72, 72);
    v16bf pa1 = load_frag_a(sP + (w * 16) * 72 + 32, 72);
#pragma unroll
    for (int ni = 0; ni < 4; ++ni) {
      v16bf vb0 = load_frag_b(sVT + (ni * 16) * 72, 72);
      v16bf vb1 = load_frag_b(sVT + (ni * 16) * 72 + 32, 72);
      o[ni] = __builtin_amdgcn_wmma_f32_16x16x32_bf16(false, pa0, false, vb0,
                                                      (short)0, o[ni], false, false);
      o[ni] = __builtin_amdgcn_wmma_f32_16x16x32_bf16(false, pa1, false, vb1,
                                                      (short)0, o[ni], false, false);
    }
  }

  // epilogue: O / l -> Y[b][t][h][d] (row-major M=B*T, K=C for proj GEMM)
#pragma unroll
  for (int ni = 0; ni < 4; ++ni) {
    int r0 = w * 16 + 8 * half;
    bf16* py = Yg + (((size_t)b * T_ + qt * 64 + r0) * NH_ + h) * HD_ + (ni * 16 + n);
#pragma unroll
    for (int e = 0; e < 8; ++e) {
      float inv = 1.0f / row_l[r0 + e];
      py[(size_t)e * NH_ * HD_] = f2bf(o[ni][e] * inv);
    }
  }
}

// ---------- launch ----------

extern "C" void kernel_launch(void* const* d_in, const int* in_sizes, int n_in,
                              void* d_out, int out_size, void* d_ws, size_t ws_size,
                              hipStream_t stream) {
  (void)in_sizes; (void)n_in; (void)out_size; (void)ws_size;
  const float* x     = (const float*)d_in[0];
  const float* Wqkv  = (const float*)d_in[1];
  const float* Wproj = (const float*)d_in[2];
  float* out = (float*)d_out;

  char* p = (char*)d_ws;
  auto take = [&](size_t bytes) -> char* {
    char* r = p;
    p += (bytes + 255) & ~(size_t)255;
    return r;
  };
  bf16*  xbf     = (bf16*)take((size_t)B_ * T_ * C_ * 2);
  bf16*  wqkvt   = (bf16*)take((size_t)C_ * NQKV_ * 2);   // [NQKV][C]
  bf16*  wprojt  = (bf16*)take((size_t)C_ * C_ * 2);      // [C][C] transposed
  float* qkvf    = (float*)take((size_t)B_ * T_ * NQKV_ * 4);
  bf16*  Qb      = (bf16*)take((size_t)B_ * NH_ * T_ * HD_ * 2);
  bf16*  Kb      = (bf16*)take((size_t)B_ * NKV_ * T_ * HD_ * 2);
  bf16*  Vtb     = (bf16*)take((size_t)B_ * NKV_ * HD_ * T_ * 2);  // [b][h][d][t]
  bf16*  Yb      = (bf16*)take((size_t)B_ * T_ * C_ * 2);

  int n1 = B_ * T_ * C_;
  cvt_f32_bf16<<<(n1 + 255) / 256, 256, 0, stream>>>(x, xbf, n1);
  int n2 = C_ * NQKV_;
  cvt_transpose_bf16<<<(n2 + 255) / 256, 256, 0, stream>>>(Wqkv, wqkvt, C_, NQKV_);
  int n3 = C_ * C_;
  cvt_transpose_bf16<<<(n3 + 255) / 256, 256, 0, stream>>>(Wproj, wprojt, C_, C_);

  // qkv = x @ W_qkv  (fp32 out for RoPE precision)
  gemm_bf16<<<dim3(NQKV_ / 128, (B_ * T_) / 128), 256, 0, stream>>>(
      xbf, wqkvt, qkvf, B_ * T_, NQKV_, C_);

  rope_scatter<<<(B_ * T_ * (NQKV_ / 2) + 255) / 256, 256, 0, stream>>>(
      qkvf, Qb, Kb, Vtb);

  attn_flash<<<dim3(T_ / 64, NH_, B_), 128, 0, stream>>>(Qb, Kb, Vtb, Yb);

  // out = Y @ W_proj (fp32 result)
  gemm_bf16<<<dim3(C_ / 128, (B_ * T_) / 128), 256, 0, stream>>>(
      Yb, wprojt, out, B_ * T_, C_, C_);
}
